// PredicateTensorModel_36472862277771
// MI455X (gfx1250) — matrix operand order repaired and checked by reading, not compiled, not run on hardware
//
#include <hip/hip_runtime.h>
#include <hip/hip_bf16.h>
#include <stdint.h>

#define D 256
#define B 512
#define NTOK 8

typedef __attribute__((ext_vector_type(16))) _Float16     v16h;
typedef __attribute__((ext_vector_type(8)))  float        v8f;
typedef __attribute__((ext_vector_type(4)))  float        f32x4;
typedef __attribute__((ext_vector_type(4)))  unsigned int u32x4;
typedef __attribute__((ext_vector_type(8)))  int          i32x8;
typedef __attribute__((ext_vector_type(4)))  int          i32x4;

union FragU { u32x4 q[2]; v16h v; };

// ---------------------------------------------------------------------------
// Stage 1: weighted embedding aggregation (gather over N=8 tokens).
// verb/subj -> f16 row-major (WMMA A-operands); obj -> f32 TRANSPOSED [p][c].
// ---------------------------------------------------------------------------
__global__ void agg_kernel(const int* __restrict__ sid, const float* __restrict__ sw,
                           const int* __restrict__ vid, const float* __restrict__ vw,
                           const int* __restrict__ oid, const float* __restrict__ ow,
                           const float* __restrict__ emb,
                           _Float16* __restrict__ verb_h, _Float16* __restrict__ subj_h,
                           float* __restrict__ objT) {
  const int c = blockIdx.x;   // 0..B-1
  const int d = threadIdx.x;  // 0..D-1
  float s = 0.f, v = 0.f, o = 0.f;
#pragma unroll
  for (int n = 0; n < NTOK; ++n) {
    s += sw[c * NTOK + n] * emb[(size_t)sid[c * NTOK + n] * D + d];
    v += vw[c * NTOK + n] * emb[(size_t)vid[c * NTOK + n] * D + d];
    o += ow[c * NTOK + n] * emb[(size_t)oid[c * NTOK + n] * D + d];
  }
  verb_h[c * D + d]       = (_Float16)v;
  subj_h[c * D + d]       = (_Float16)s;
  objT[(size_t)d * B + c] = o;        // [p][c] so the hot loop reads contiguously
}

// ---------------------------------------------------------------------------
// Stage 2: one-time transpose+convert  w[i][p][q] -> wT[p][q][i]  (f32 -> f16).
// Makes every (p, q-tile) slab a contiguous 16KB block: perfect TDM shape,
// and B-fragments become 16 contiguous halves per lane.
// ---------------------------------------------------------------------------
__global__ __launch_bounds__(256) void transpose_convert_kernel(
    const float* __restrict__ W, const float* __restrict__ U,
    _Float16* __restrict__ WT, _Float16* __restrict__ UT) {
  __shared__ float lds_t[16 * D];
  const int p  = blockIdx.x;       // 0..255
  const int q0 = blockIdx.y * 16;  // q block
  const float* __restrict__ src = blockIdx.z ? U : W;
  _Float16* __restrict__    dst = blockIdx.z ? UT : WT;
  const int t  = threadIdx.x;
  const int qi = t & 15;
  const int ib = t >> 4;
#pragma unroll
  for (int it = 0; it < 16; ++it) {        // coalesced reads along q
    const int i = ib + 16 * it;
    lds_t[qi * D + i] = src[(size_t)i * (D * D) + (size_t)p * D + q0 + qi];
  }
  __syncthreads();
#pragma unroll
  for (int jt = 0; jt < 16; ++jt) {        // coalesced 512B writes along i
    dst[((size_t)p * D + q0 + jt) * D + t] = (_Float16)lds_t[jt * D + t];
  }
}

// ---------------------------------------------------------------------------
// TDM: issue one contiguous 16KB (8192 x f16) tile load, global -> LDS.
// D# per CDNA5 ISA ch.8: group0 {count,lds_addr,global_addr,type=2},
// group1 {data_size=2B, tensor_dim0=tile_dim0=8192, dim0_stride=8192}.
// This toolchain's builtin takes 6 args (extra i32x8 group; zero-filled).
// ---------------------------------------------------------------------------
__device__ __forceinline__ void tdm_load_slab(const _Float16* gsrc, _Float16* ldst) {
  const unsigned lds_off = (unsigned)(uintptr_t)ldst;          // LDS aperture low 32b
  const unsigned long long ga = (unsigned long long)(uintptr_t)gsrc;
  u32x4 g0;
  g0[0] = 1u;                                                  // count=1 (valid, user)
  g0[1] = lds_off;                                             // lds_addr
  g0[2] = (unsigned)(ga & 0xFFFFFFFFu);                        // global_addr[31:0]
  g0[3] = (unsigned)((ga >> 32) & 0x1FFFFFFu) | 0x80000000u;   // ga[56:32] | type=2
  i32x8 g1;
  g1[0] = 0x00010000;             // workgroup_mask=0, data_size=1 (2 bytes)
  g1[1] = (int)(8192u << 16);     // tensor_dim0[15:0]=8192 (bits [63:48])
  g1[2] = (int)(1u << 16);        // tensor_dim1=1
  g1[3] = (int)(8192u << 16);     // tile_dim0=8192 (bits [127:112])
  g1[4] = 0;                      // tile_dim1=0, tile_dim2=0 (unused)
  g1[5] = 8192;                   // tensor_dim0_stride[31:0]
  g1[6] = 0;
  g1[7] = 0;
  i32x4 g2 = {0, 0, 0, 0};
  i32x4 g3 = {0, 0, 0, 0};
  i32x8 g4 = {0, 0, 0, 0, 0, 0, 0, 0};
  __builtin_amdgcn_tensor_load_to_lds(g0, g1, g2, g3, g4, 0);
}

// ---------------------------------------------------------------------------
// Stage 3: fused final[c,q] = sum_p obj[c,p] * (verb@w)[c,p,q] * (subj@u)[c,p,q]
// TDM double-buffers 16KB w/u slabs into LDS; 16 v_wmma per p per wave.
// ---------------------------------------------------------------------------
__global__ __launch_bounds__(128) void fused_kernel(
    const _Float16* __restrict__ verb_h, const _Float16* __restrict__ subj_h,
    const float* __restrict__ objT, const _Float16* __restrict__ WT,
    const _Float16* __restrict__ UT, float* __restrict__ out) {
  __shared__ _Float16 lds_w[2][32 * D];   // [buf][q_local][i], 16KB each
  __shared__ _Float16 lds_u[2][32 * D];

  const int t     = threadIdx.x;
  const int lane  = t & 31;
  const int wv    = t >> 5;        // wave 0..3
  const int half  = lane >> 4;
  const int nl    = lane & 15;
  const int c_sub = wv >> 1;       // 0..1
  const int q_sub = wv & 1;        // 0..1
  const int c0b   = blockIdx.x * 32;
  const int q0b   = blockIdx.y * 32;

  // p-invariant A-fragments (verb & subj) pinned in registers.
  v16h av[8], asb[8];
  {
    const _Float16* vr = verb_h + (size_t)(c0b + c_sub * 16 + nl) * D;
    const _Float16* sr = subj_h + (size_t)(c0b + c_sub * 16 + nl) * D;
    const int koff = half * 16;
#pragma unroll
    for (int ks = 0; ks < 8; ++ks) {
      FragU fa, fs;
      fa.q[0] = *(const u32x4*)(vr + ks * 32 + koff);
      fa.q[1] = *(const u32x4*)(vr + ks * 32 + 8 + koff);
      fs.q[0] = *(const u32x4*)(sr + ks * 32 + koff);
      fs.q[1] = *(const u32x4*)(sr + ks * 32 + 8 + koff);
      av[ks]  = fa.v;
      asb[ks] = fs.v;
    }
  }

  float facc[8];
#pragma unroll
  for (int r = 0; r < 8; ++r) facc[r] = 0.f;

  // prologue: TDM-fill buffer 0 with the p=0 slabs
  if (t < 32) {
    tdm_load_slab(WT + (size_t)q0b * D, &lds_w[0][0]);
    tdm_load_slab(UT + (size_t)q0b * D, &lds_u[0][0]);
    __builtin_amdgcn_s_wait_tensorcnt(0);
  }
  __syncthreads();

  const int qrow = q_sub * 16 + nl;
  for (int p = 0; p < D; ++p) {
    const int cur = p & 1;
    const int nxt = cur ^ 1;

    // async prefetch of slab p+1 while computing slab p
    if ((p + 1 < D) && (t < 32)) {
      const size_t slab = ((size_t)(p + 1) * D + q0b) * D;
      tdm_load_slab(WT + slab, &lds_w[nxt][0]);
      tdm_load_slab(UT + slab, &lds_u[nxt][0]);
    }

    // two K=256 WMMA chains (A = verb@w, US = subj@u) for 16x16 (c,q) sub-tile
    v8f accA = {};
    v8f accU = {};
#pragma unroll
    for (int ks = 0; ks < 8; ++ks) {
      FragU bw, bu;
      const _Float16* wl = &lds_w[cur][qrow * D + ks * 32 + half * 16];
      const _Float16* ul = &lds_u[cur][qrow * D + ks * 32 + half * 16];
      bw.q[0] = *(const u32x4*)(wl);
      bw.q[1] = *(const u32x4*)(wl + 8);
      bu.q[0] = *(const u32x4*)(ul);
      bu.q[1] = *(const u32x4*)(ul + 8);
      accA = __builtin_amdgcn_wmma_f32_16x16x32_f16(false, av[ks],  false, bw.v,
                                                    (short)0, accA, false, false);
      accU = __builtin_amdgcn_wmma_f32_16x16x32_f16(false, asb[ks], false, bu.v,
                                                    (short)0, accU, false, false);
    }

    // fold obj[c,p] * A * US into the p-reduction. objT is [p][c]: the 8 rows
    // this lane owns are 8 consecutive floats -> two 16B loads (L2 resident).
    {
      const float* ob = objT + (size_t)p * B + c0b + c_sub * 16 + half * 8;
      const f32x4 o0 = *(const f32x4*)(ob);
      const f32x4 o1 = *(const f32x4*)(ob + 4);
#pragma unroll
      for (int r = 0; r < 4; ++r) {
        facc[r]     += o0[r] * accA[r]     * accU[r];
        facc[r + 4] += o1[r] * accA[r + 4] * accU[r + 4];
      }
    }

    // drain TDM (wave 0) then release buffers to everyone
    if (t < 32) __builtin_amdgcn_s_wait_tensorcnt(0);
    __syncthreads();
  }

  // write final (B x D), f32
#pragma unroll
  for (int r = 0; r < 8; ++r) {
    out[(size_t)(c0b + c_sub * 16 + half * 8 + r) * D + (q0b + q_sub * 16 + nl)] =
        facc[r];
  }
}

extern "C" void kernel_launch(void* const* d_in, const int* in_sizes, int n_in,
                              void* d_out, int out_size, void* d_ws, size_t ws_size,
                              hipStream_t stream) {
  const int*   sid = (const int*)d_in[0];
  const float* sw  = (const float*)d_in[1];
  const int*   vid = (const int*)d_in[2];
  const float* vw  = (const float*)d_in[3];
  const int*   oid = (const int*)d_in[4];
  const float* ow  = (const float*)d_in[5];
  const float* emb = (const float*)d_in[6];
  const float* W   = (const float*)d_in[7];
  const float* U   = (const float*)d_in[8];

  char* ws = (char*)d_ws;
  _Float16* verb_h = (_Float16*)ws;                            // B*D f16   (256KB)
  _Float16* subj_h = (_Float16*)(ws + (size_t)B * D * 2);      // B*D f16   (256KB)
  float*    objT   = (float*)(ws + (size_t)B * D * 4);         // D*B f32   (512KB)
  _Float16* WT     = (_Float16*)(ws + (size_t)B * D * 8);      // D^3 f16   (32MB)
  _Float16* UT     = WT + (size_t)D * D * D;                   // D^3 f16   (32MB)

  agg_kernel<<<B, D, 0, stream>>>(sid, sw, vid, vw, oid, ow, emb,
                                  verb_h, subj_h, objT);

  dim3 tg(D, 16, 2);  // (p, q-block, {w,u})
  transpose_convert_kernel<<<tg, 256, 0, stream>>>(W, U, WT, UT);

  dim3 grid(B / 32, D / 32);  // (c-tiles, q-tiles)
  fused_kernel<<<grid, 128, 0, stream>>>(verb_h, subj_h, objT, WT, UT,
                                         (float*)d_out);
}
